// AttentionBlock_833223655801
// MI455X (gfx1250) — matrix operand order repaired and checked by reading, not compiled, not run on hardware
//
#include <hip/hip_runtime.h>
#include <hip/hip_bf16.h>
#include <stdint.h>

// Problem constants (match reference setup_inputs)
#define BATCH   4
#define CH      256
#define GROUPS  8
#define CG      (CH / GROUPS)   // 32
#define TOK     4096            // H*W
#define UNITS   256
#define GN_EPS  1e-3f
#define ATT_SCALE 0.0625f       // 256^-0.5

typedef __attribute__((ext_vector_type(16))) __bf16       v16bf;
typedef __attribute__((ext_vector_type(8)))  float        v8f;
typedef __attribute__((ext_vector_type(4)))  unsigned int u32x4;

union Frag { u32x4 u[2]; v16bf v; };   // 32 bytes = one 16-bit WMMA A/B fragment per lane

static __device__ inline v8f wmma_bf16(v16bf a, v16bf b, v8f c) {
  return __builtin_amdgcn_wmma_f32_16x16x32_bf16(false, a, false, b, (short)0, c,
                                                 false, false);
}

static __device__ inline float hmax16(float v) {
  v = fmaxf(v, __shfl_xor(v, 1, 16));
  v = fmaxf(v, __shfl_xor(v, 2, 16));
  v = fmaxf(v, __shfl_xor(v, 4, 16));
  v = fmaxf(v, __shfl_xor(v, 8, 16));
  return v;
}
static __device__ inline float hsum16(float v) {
  v += __shfl_xor(v, 1, 16);
  v += __shfl_xor(v, 2, 16);
  v += __shfl_xor(v, 4, 16);
  v += __shfl_xor(v, 8, 16);
  return v;
}

// ---------------------------------------------------------------------------
// Kernel 1: per-(batch, group) mean / rstd.  32 blocks x 256 threads.
// ---------------------------------------------------------------------------
__global__ void gn_stats_kernel(const float* __restrict__ x, float* __restrict__ stats) {
  const int bg = blockIdx.x;            // 0..31
  const int b = bg >> 3, g = bg & 7;
  const float* base = x + (size_t)b * TOK * CH + g * CG;
  float s = 0.f, s2 = 0.f;
  for (int i = threadIdx.x; i < TOK * CG; i += 256) {
    const int hw = i >> 5, j = i & 31;
    const float v = base[(size_t)hw * CH + j];
    s += v; s2 += v * v;
  }
  __shared__ float sh[256], sh2[256];
  sh[threadIdx.x] = s; sh2[threadIdx.x] = s2;
  __syncthreads();
  for (int off = 128; off > 0; off >>= 1) {
    if (threadIdx.x < off) {
      sh[threadIdx.x]  += sh[threadIdx.x + off];
      sh2[threadIdx.x] += sh2[threadIdx.x + off];
    }
    __syncthreads();
  }
  if (threadIdx.x == 0) {
    const float inv_n = 1.0f / (float)(TOK * CG);
    const float mean = sh[0] * inv_n;
    const float var  = sh2[0] * inv_n - mean * mean;
    stats[bg * 2 + 0] = mean;
    stats[bg * 2 + 1] = rsqrtf(var + GN_EPS);
  }
}

// ---------------------------------------------------------------------------
// Kernel 2: apply group norm -> xn (f32, residual) and xnb (bf16, GEMM input)
// ---------------------------------------------------------------------------
__global__ void gn_apply_kernel(const float* __restrict__ x,
                                const float* __restrict__ gamma,
                                const float* __restrict__ beta,
                                const float* __restrict__ stats,
                                float* __restrict__ xn, __bf16* __restrict__ xnb) {
  const size_t i = (size_t)blockIdx.x * 256 + threadIdx.x;   // < 4*4096*256 = 2^22
  const int c = (int)(i & (CH - 1));
  const int b = (int)(i >> 20);                              // TOK*CH = 2^20
  const int g = c >> 5;
  const float mean = stats[(b * 8 + g) * 2 + 0];
  const float rstd = stats[(b * 8 + g) * 2 + 1];
  const float v = (x[i] - mean) * rstd * gamma[c] + beta[c];
  xn[i]  = v;
  xnb[i] = (__bf16)v;
}

// ---------------------------------------------------------------------------
// Kernel 3: transpose+convert Wq/Wk/Wv/Wp to bf16 [N][K] for B-fragments
// ---------------------------------------------------------------------------
__global__ void wconv_kernel(const float* __restrict__ Wq, const float* __restrict__ Wk,
                             const float* __restrict__ Wv, const float* __restrict__ Wp,
                             __bf16* __restrict__ WT) {
  const int i = blockIdx.x * 256 + threadIdx.x;              // < 4*65536
  const int m = i >> 16;
  const int n = (i >> 8) & 255;
  const int k = i & 255;
  const float* W = (m == 0) ? Wq : (m == 1) ? Wk : (m == 2) ? Wv : Wp;
  WT[i] = (__bf16)W[k * 256 + n];                            // WT[m][n][k] = W[k][n]
}

// ---------------------------------------------------------------------------
// Kernel 4: QKV GEMM.  One wave per 16(t) x 64(c) tile of one of {q,k,v}.
// q,k stored row-major [T][C] bf16; v stored transposed [C][T] bf16.
// grid = (TOK/16, 4, BATCH*3), block = 32
// ---------------------------------------------------------------------------
__global__ void qkv_gemm_kernel(const __bf16* __restrict__ xnb,
                                const __bf16* __restrict__ WT,
                                const float* __restrict__ bq,
                                const float* __restrict__ bk,
                                const float* __restrict__ bv,
                                __bf16* __restrict__ qb, __bf16* __restrict__ kb,
                                __bf16* __restrict__ vtb) {
  const int tt = blockIdx.x;          // token tile
  const int cs = blockIdx.y;          // 64-col slice
  const int bz = blockIdx.z;
  const int b = bz & 3, mode = bz >> 2;   // 0=q 1=k 2=v
  const int lane = threadIdx.x, hi = lane >> 4, ln = lane & 15;

  const __bf16* A  = xnb + (size_t)b * TOK * CH + (size_t)(tt * 16 + ln) * CH;
  const __bf16* Wt = WT + (size_t)mode * 65536;

  v8f acc[4] = {};
#pragma unroll
  for (int k0 = 0; k0 < CH; k0 += 32) {
    Frag a;
    const __bf16* ap = A + k0 + hi * 16;
    a.u[0] = *(const u32x4*)ap;
    a.u[1] = *(const u32x4*)(ap + 8);
#pragma unroll
    for (int j = 0; j < 4; j++) {
      const int n = cs * 64 + j * 16 + ln;
      const __bf16* wp = Wt + (size_t)n * CH + k0 + hi * 16;
      Frag bfr;
      bfr.u[0] = *(const u32x4*)wp;
      bfr.u[1] = *(const u32x4*)(wp + 8);
      acc[j] = wmma_bf16(a.v, bfr.v, acc[j]);
    }
  }

  const float* bias = (mode == 0) ? bq : (mode == 1) ? bk : bv;
  if (mode < 2) {
    __bf16* O = ((mode == 0) ? qb : kb) + (size_t)b * TOK * CH;
#pragma unroll
    for (int j = 0; j < 4; j++)
#pragma unroll
      for (int r = 0; r < 8; r++) {
        const int t = tt * 16 + r + hi * 8;
        const int c = cs * 64 + j * 16 + ln;
        O[(size_t)t * CH + c] = (__bf16)(acc[j][r] + bias[c]);
      }
  } else {
    __bf16* O = vtb + (size_t)b * CH * TOK;   // transposed [C][T]
#pragma unroll
    for (int j = 0; j < 4; j++)
#pragma unroll
      for (int r = 0; r < 8; r++) {
        const int t = tt * 16 + r + hi * 8;
        const int c = cs * 64 + j * 16 + ln;
        O[(size_t)c * TOK + t] = (__bf16)(acc[j][r] + bias[c]);
      }
  }
}

// ---------------------------------------------------------------------------
// Kernel 5: flash attention.  One wave per (batch, 16-row q tile, 64-col slice).
// Streams all 4096 keys in steps of 32, online softmax, PV accumulate.
// grid = (TOK/16, 4, BATCH), block = 32
// ---------------------------------------------------------------------------
__global__ void attn_kernel(const __bf16* __restrict__ qb,
                            const __bf16* __restrict__ kb,
                            const __bf16* __restrict__ vtb,
                            __bf16* __restrict__ ab) {
  const int tt = blockIdx.x, cs = blockIdx.y, b = blockIdx.z;
  const int lane = threadIdx.x, hi = lane >> 4, ln = lane & 15;

  __shared__ __bf16 P[16][32];   // softmaxed score tile, D-layout -> A-layout bounce

  // Q fragments for this 16-row tile (kept in registers: 64 VGPRs)
  Frag qf[8];
  const __bf16* Q = qb + (size_t)b * TOK * CH + (size_t)(tt * 16 + ln) * CH;
#pragma unroll
  for (int i = 0; i < 8; i++) {
    const __bf16* p = Q + i * 32 + hi * 16;
    qf[i].u[0] = *(const u32x4*)p;
    qf[i].u[1] = *(const u32x4*)(p + 8);
  }

  const __bf16* K  = kb  + (size_t)b * TOK * CH;
  const __bf16* Vt = vtb + (size_t)b * CH * TOK;

  float mrow[8], lrow[8];
#pragma unroll
  for (int r = 0; r < 8; r++) { mrow[r] = -3.0e38f; lrow[r] = 0.f; }
  v8f acc[4] = {};

  for (int s0 = 0; s0 < TOK; s0 += 32) {
    // ---- S tile: 16 rows x 32 keys (two 16x16 WMMA accumulators) ----
    v8f sl = {}, sr = {};
#pragma unroll
    for (int i = 0; i < 8; i++) {
      const __bf16* kl = K + (size_t)(s0 + ln) * CH + i * 32 + hi * 16;
      Frag fl;
      fl.u[0] = *(const u32x4*)kl;
      fl.u[1] = *(const u32x4*)(kl + 8);
      sl = wmma_bf16(qf[i].v, fl.v, sl);
      const __bf16* kr = kl + 16 * CH;
      Frag fr;
      fr.u[0] = *(const u32x4*)kr;
      fr.u[1] = *(const u32x4*)(kr + 8);
      sr = wmma_bf16(qf[i].v, fr.v, sr);
    }

    // ---- online softmax update (row r lives across a 16-lane half) ----
#pragma unroll
    for (int r = 0; r < 8; r++) {
      const float a  = sl[r] * ATT_SCALE;
      const float c2 = sr[r] * ATT_SCALE;
      const float tmax = hmax16(fmaxf(a, c2));
      const float mnew = fmaxf(mrow[r], tmax);
      const float corr = __expf(mrow[r] - mnew);
      const float pl = __expf(a - mnew);
      const float pr = __expf(c2 - mnew);
      const float psum = hsum16(pl + pr);
      lrow[r] = lrow[r] * corr + psum;
      mrow[r] = mnew;
      acc[0][r] *= corr; acc[1][r] *= corr; acc[2][r] *= corr; acc[3][r] *= corr;
      P[r + hi * 8][ln]      = (__bf16)pl;
      P[r + hi * 8][ln + 16] = (__bf16)pr;
    }
    __syncthreads();   // single-wave WG: cheap fence ordering LDS store->load

    // ---- re-read P in A-fragment layout ----
    Frag pf;
    const __bf16* pp = &P[ln][hi * 16];
    pf.u[0] = *(const u32x4*)pp;
    pf.u[1] = *(const u32x4*)(pp + 8);

    // ---- PV: accumulate 16 x 64 output slice ----
#pragma unroll
    for (int j = 0; j < 4; j++) {
      const __bf16* vp = Vt + (size_t)(cs * 64 + j * 16 + ln) * TOK + s0 + hi * 16;
      Frag vf;
      vf.u[0] = *(const u32x4*)vp;
      vf.u[1] = *(const u32x4*)(vp + 8);
      acc[j] = wmma_bf16(pf.v, vf.v, acc[j]);
    }
    __syncthreads();
  }

  // ---- normalize and store attention output (bf16, row-major) ----
  __bf16* O = ab + (size_t)b * TOK * CH;
#pragma unroll
  for (int r = 0; r < 8; r++) {
    const float inv_l = 1.0f / lrow[r];
    const int t = tt * 16 + r + hi * 8;
#pragma unroll
    for (int j = 0; j < 4; j++) {
      const int c = cs * 64 + j * 16 + ln;
      O[(size_t)t * CH + c] = (__bf16)(acc[j][r] * inv_l);
    }
  }
}

// ---------------------------------------------------------------------------
// Kernel 6: final projection + bias + residual -> f32 output
// grid = (TOK/16, 4, BATCH), block = 32
// ---------------------------------------------------------------------------
__global__ void proj_kernel(const __bf16* __restrict__ ab,
                            const __bf16* __restrict__ WT,   // WpT at +3*65536
                            const float* __restrict__ bp,
                            const float* __restrict__ xn,
                            float* __restrict__ out) {
  const int tt = blockIdx.x, cs = blockIdx.y, b = blockIdx.z;
  const int lane = threadIdx.x, hi = lane >> 4, ln = lane & 15;

  const __bf16* A  = ab + (size_t)b * TOK * CH + (size_t)(tt * 16 + ln) * CH;
  const __bf16* Wt = WT + (size_t)3 * 65536;

  v8f acc[4] = {};
#pragma unroll
  for (int k0 = 0; k0 < CH; k0 += 32) {
    Frag a;
    const __bf16* ap = A + k0 + hi * 16;
    a.u[0] = *(const u32x4*)ap;
    a.u[1] = *(const u32x4*)(ap + 8);
#pragma unroll
    for (int j = 0; j < 4; j++) {
      const int n = cs * 64 + j * 16 + ln;
      const __bf16* wp = Wt + (size_t)n * CH + k0 + hi * 16;
      Frag bfr;
      bfr.u[0] = *(const u32x4*)wp;
      bfr.u[1] = *(const u32x4*)(wp + 8);
      acc[j] = wmma_bf16(a.v, bfr.v, acc[j]);
    }
  }

#pragma unroll
  for (int j = 0; j < 4; j++)
#pragma unroll
    for (int r = 0; r < 8; r++) {
      const int t = tt * 16 + r + hi * 8;
      const int c = cs * 64 + j * 16 + ln;
      const size_t idx = (size_t)b * TOK * UNITS + (size_t)t * UNITS + c;
      out[idx] = xn[idx] + acc[j][r] + bp[c];
    }
}

// ---------------------------------------------------------------------------
// Host launch
// ---------------------------------------------------------------------------
extern "C" void kernel_launch(void* const* d_in, const int* in_sizes, int n_in,
                              void* d_out, int out_size, void* d_ws, size_t ws_size,
                              hipStream_t stream) {
  const float* x     = (const float*)d_in[0];
  const float* gamma = (const float*)d_in[1];
  const float* beta  = (const float*)d_in[2];
  const float* Wq    = (const float*)d_in[3];
  const float* bq    = (const float*)d_in[4];
  const float* Wk    = (const float*)d_in[5];
  const float* bk    = (const float*)d_in[6];
  const float* Wv    = (const float*)d_in[7];
  const float* bv    = (const float*)d_in[8];
  const float* Wp    = (const float*)d_in[9];
  const float* bp    = (const float*)d_in[10];
  float* out = (float*)d_out;

  // Workspace layout (bytes)
  char* ws = (char*)d_ws;
  const size_t SZ_F32  = (size_t)BATCH * TOK * CH * sizeof(float);    // 16 MB
  const size_t SZ_BF16 = (size_t)BATCH * TOK * CH * sizeof(__bf16);   //  8 MB
  float*  stats = (float*)(ws + 0);
  float*  xn    = (float*)(ws + 4096);
  __bf16* xnb   = (__bf16*)(ws + 4096 + SZ_F32);
  __bf16* qb    = (__bf16*)(ws + 4096 + SZ_F32 + 1 * SZ_BF16);
  __bf16* kb    = (__bf16*)(ws + 4096 + SZ_F32 + 2 * SZ_BF16);
  __bf16* vtb   = (__bf16*)(ws + 4096 + SZ_F32 + 3 * SZ_BF16);
  __bf16* ab    = (__bf16*)(ws + 4096 + SZ_F32 + 4 * SZ_BF16);
  __bf16* WT    = (__bf16*)(ws + 4096 + SZ_F32 + 5 * SZ_BF16);        // 4 x 256x256 bf16

  gn_stats_kernel<<<32, 256, 0, stream>>>(x, stats);
  gn_apply_kernel<<<(BATCH * TOK * CH) / 256, 256, 0, stream>>>(x, gamma, beta, stats,
                                                                xn, xnb);
  wconv_kernel<<<(4 * 256 * 256) / 256, 256, 0, stream>>>(Wq, Wk, Wv, Wp, WT);
  qkv_gemm_kernel<<<dim3(TOK / 16, 4, BATCH * 3), 32, 0, stream>>>(xnb, WT, bq, bk, bv,
                                                                   qb, kb, vtb);
  attn_kernel<<<dim3(TOK / 16, 4, BATCH), 32, 0, stream>>>(qb, kb, vtb, ab);
  proj_kernel<<<dim3(TOK / 16, 4, BATCH), 32, 0, stream>>>(ab, WT, bp, xn, out);
}
